// ResBlock_47064251630165
// MI455X (gfx1250) — matrix operand order repaired and checked by reading, not compiled, not run on hardware
//
#include <hip/hip_runtime.h>

// ---------------------------------------------------------------------------
// CDNA5 (gfx1250, wave32) GNN ResBlock:
//   y   = relu(LN( mean_agg(x) @ W0n + x @ W0r + b0 ))
//   out = relu(LN( mean_agg(y) @ W1n + y @ W1r + b1 + x @ Wlin + blin ))
// GEMMs: v_wmma_f32_16x16x32_bf16, fp32 emulated via bf16 hi/lo split
// (Ah*Bh + Ah*Bl + Al*Bh). A-tiles staged via GLOBAL_LOAD_ASYNC_TO_LDS_B128
// (ASYNCcnt path). Scatter/gather phase is L2-resident f32 atomics.
// ---------------------------------------------------------------------------

typedef __attribute__((ext_vector_type(16))) __bf16 v16bf;
typedef __attribute__((ext_vector_type(8)))  float  v8f;

#define NN   50000
#define NE   800000
#define CIN  128
#define COUT 256
#define LN_EPS 1e-5f

__device__ __forceinline__ v8f wmma_bf16(v16bf a, v16bf b, v8f c) {
  return __builtin_amdgcn_wmma_f32_16x16x32_bf16(
      /*neg_a=*/false, a, /*neg_b=*/false, b,
      /*c_mod=*/(short)0, c, /*reuse_a=*/false, /*reuse_b=*/false);
}

// Async global->LDS copy, 16 bytes per lane (ASYNCcnt-tracked).
// lds_off = wave-relative LDS byte offset (low 32 bits of generic shared addr).
__device__ __forceinline__ void async_ld_b128(unsigned lds_off, const void* gaddr) {
  asm volatile("global_load_async_to_lds_b128 %0, %1, off"
               :: "v"(lds_off), "v"(gaddr) : "memory");
}
__device__ __forceinline__ void wait_async0() {
  asm volatile("s_wait_asynccnt 0" ::: "memory");
}

// ---------------------------------------------------------------------------
// One-time weight prep: transpose to [n][k] and split fp32 -> bf16 hi/lo.
// B0 = [W0n ; W0r]            (K=256, N=256)
// B1 = [W1n ; W1r ; Wlin]     (K=640, N=256), bias1 = b1 + blin
// ---------------------------------------------------------------------------
__global__ __launch_bounds__(256) void prep_weights(
    const float* __restrict__ w0n, const float* __restrict__ w0r,
    const float* __restrict__ w1n, const float* __restrict__ w1r,
    const float* __restrict__ linw,
    const float* __restrict__ b0, const float* __restrict__ b1,
    const float* __restrict__ linb,
    __bf16* __restrict__ B0hi, __bf16* __restrict__ B0lo,
    __bf16* __restrict__ B1hi, __bf16* __restrict__ B1lo,
    float* __restrict__ bias0, float* __restrict__ bias1)
{
  int idx = blockIdx.x * blockDim.x + threadIdx.x;
  if (idx < COUT * 256) {               // B0: 256 cols x K=256
    int n = idx >> 8, k = idx & 255;
    float v = (k < CIN) ? w0n[k * COUT + n] : w0r[(k - CIN) * COUT + n];
    __bf16 h = (__bf16)v;
    B0hi[n * 256 + k] = h;
    B0lo[n * 256 + k] = (__bf16)(v - (float)h);
  }
  if (idx < COUT * 640) {               // B1: 256 cols x K=640
    int n = idx / 640, k = idx - n * 640;
    float v = (k < 256) ? w1n[k * COUT + n]
            : (k < 512) ? w1r[(k - 256) * COUT + n]
                        : linw[(k - 512) * COUT + n];
    __bf16 h = (__bf16)v;
    B1hi[n * 640 + k] = h;
    B1lo[n * 640 + k] = (__bf16)(v - (float)h);
  }
  if (idx < COUT) {
    bias0[idx] = b0[idx];
    bias1[idx] = b1[idx] + linb[idx];
  }
}

// ---------------------------------------------------------------------------
// Edge kernels
// ---------------------------------------------------------------------------
__global__ __launch_bounds__(256) void deg_kernel(
    const int* __restrict__ dst, float* __restrict__ deg)
{
  int e = blockIdx.x * blockDim.x + threadIdx.x;
  if (e < NE) atomicAdd(&deg[dst[e]], 1.0f);
}

__global__ __launch_bounds__(256) void invdeg_kernel(
    const float* __restrict__ deg, float* __restrict__ inv)
{
  int i = blockIdx.x * blockDim.x + threadIdx.x;
  if (i < NN) inv[i] = 1.0f / fmaxf(deg[i], 1.0f);
}

// One wave32 per edge; lanes stride the channel dim (coalesced 128B bursts).
// Edge index is wave-uniform -> force scalar loads via readfirstlane.
template<int C>
__global__ __launch_bounds__(256) void scatter_kernel(
    const float* __restrict__ feat, const int* __restrict__ src,
    const int* __restrict__ dst, float* __restrict__ agg)
{
  int wid  = (blockIdx.x * blockDim.x + threadIdx.x) >> 5;
  int lane = threadIdx.x & 31;
  if (wid >= NE) return;
  wid = __builtin_amdgcn_readfirstlane(wid);
  int s = __builtin_amdgcn_readfirstlane(src[wid]);
  int d = __builtin_amdgcn_readfirstlane(dst[wid]);
  const float* fp = feat + (size_t)s * C;
  float*       ap = agg  + (size_t)d * C;
  #pragma unroll
  for (int j = 0; j < C / 32; ++j)
    atomicAdd(&ap[lane + 32 * j], fp[lane + 32 * j]);
}

// ---------------------------------------------------------------------------
// Fused GEMM + bias + per-node LayerNorm + ReLU.
// A = [A0 | A1 | A2] async-copied into LDS (16 rows x KTOT fp32);
// A0 rows get inverse-degree scaling at fragment-build time.
// B pre-transposed [n][KTOT] bf16 hi/lo. 8 waves x (two 16x16 tiles) = N=256.
// ---------------------------------------------------------------------------
template<int KTOT>
__global__ __launch_bounds__(256) void gemm_ln_relu(
    const float* __restrict__ A0, const float* __restrict__ scale0, int K0,
    const float* __restrict__ A1, int K1,
    const float* __restrict__ A2, int K2,
    const __bf16* __restrict__ Bhi, const __bf16* __restrict__ Blo,
    const float* __restrict__ bias,
    const float* __restrict__ lng, const float* __restrict__ lnb,
    float* __restrict__ outp)
{
  __shared__ float As[16 * KTOT];
  const int tid  = threadIdx.x;
  const int row0 = blockIdx.x * 16;

  // --- Async-stage the A tile: 4*KTOT 16-byte chunks, one per lane/iter ---
  // LDS tile layout: row r holds [A0 row | A1 row | A2 row] (KTOT floats).
  for (int c = tid; c < 4 * KTOT; c += 256) {
    int fo = c * 4;                 // float offset within tile
    int r  = fo / KTOT;
    int k  = fo - r * KTOT;
    const float* gp;
    if (k < K0)           gp = A0 + (size_t)(row0 + r) * K0 + k;
    else if (k < K0 + K1) gp = A1 + (size_t)(row0 + r) * K1 + (k - K0);
    else                  gp = A2 + (size_t)(row0 + r) * K2 + (k - K0 - K1);
    unsigned lds_off = (unsigned)(unsigned long long)(&As[fo]);
    async_ld_b128(lds_off, gp);
  }
  wait_async0();
  __syncthreads();

  const int wave = tid >> 5;
  const int lane = tid & 31;
  const int half = lane >> 4;     // K-half: k = 16*half + i (contiguous order used here)
  const int lr   = lane & 15;

  const int n0   = wave * 32;     // two adjacent 16-wide column tiles
  const int colA = n0 + lr, colB = n0 + 16 + lr;
  const float rowScale = scale0[row0 + lr];   // applies to A0 k-range only

  v8f acc0 = {}; v8f acc1 = {};

  for (int k0 = 0; k0 < KTOT; k0 += 32) {
    const float sc = (k0 < K0) ? rowScale : 1.0f;   // K0 % 32 == 0
    // A fragment: lane holds row lr; elems i -> K = k0+16*half+i
    v16bf ahi, alo;
    const float* ap = &As[lr * KTOT + k0 + 16 * half];
    #pragma unroll
    for (int i = 0; i < 8; ++i) {
      float v = ap[i] * sc;
      __bf16 h = (__bf16)v;
      ahi[i] = h; alo[i] = (__bf16)(v - (float)h);
    }
    #pragma unroll
    for (int i = 0; i < 8; ++i) {
      float v = ap[8 + i] * sc;
      __bf16 h = (__bf16)v;
      ahi[8 + i] = h; alo[8 + i] = (__bf16)(v - (float)h);
    }
    // B fragments: lane holds column n; elems i -> K = k0+16*half+i
    const __bf16* bh0p = Bhi + (size_t)colA * KTOT + k0 + 16 * half;
    const __bf16* bl0p = Blo + (size_t)colA * KTOT + k0 + 16 * half;
    const __bf16* bh1p = Bhi + (size_t)colB * KTOT + k0 + 16 * half;
    const __bf16* bl1p = Blo + (size_t)colB * KTOT + k0 + 16 * half;
    v16bf bh0, bl0, bh1, bl1;
    #pragma unroll
    for (int i = 0; i < 16; ++i) {
      bh0[i] = bh0p[i]; bl0[i] = bl0p[i];
      bh1[i] = bh1p[i]; bl1[i] = bl1p[i];
    }
    // fp32-emulating 3-term bf16 WMMA
    acc0 = wmma_bf16(ahi, bh0, acc0);
    acc0 = wmma_bf16(ahi, bl0, acc0);
    acc0 = wmma_bf16(alo, bh0, acc0);
    acc1 = wmma_bf16(ahi, bh1, acc1);
    acc1 = wmma_bf16(ahi, bl1, acc1);
    acc1 = wmma_bf16(alo, bh1, acc1);
  }

  // Epilogue: bias add, park tile in LDS, per-row LayerNorm + ReLU
  float bA = bias[colA], bB = bias[colB];
  __syncthreads();                 // all waves done reading As
  float* Cs = As;                  // reuse LDS as 16x256 fp32 tile
  #pragma unroll
  for (int i = 0; i < 8; ++i) {    // C layout: lane n=col, VGPR i -> M=i+8*half
    Cs[(i + 8 * half) * 256 + colA] = acc0[i] + bA;
    Cs[(i + 8 * half) * 256 + colB] = acc1[i] + bB;
  }
  __syncthreads();

  #pragma unroll
  for (int rr = 0; rr < 2; ++rr) { // 8 waves x 2 rows = 16 rows
    int r = wave * 2 + rr;
    float s = 0.f, sq = 0.f;
    #pragma unroll
    for (int j = 0; j < 8; ++j) {
      float v = Cs[r * 256 + lane + 32 * j];
      s += v; sq += v * v;
    }
    #pragma unroll
    for (int off = 16; off > 0; off >>= 1) {   // wave32 reduction
      s  += __shfl_xor(s,  off, 32);
      sq += __shfl_xor(sq, off, 32);
    }
    float mean = s * (1.0f / 256.0f);
    float var  = sq * (1.0f / 256.0f) - mean * mean;
    float rstd = rsqrtf(var + LN_EPS);
    #pragma unroll
    for (int j = 0; j < 8; ++j) {
      int n = lane + 32 * j;
      float v = (Cs[r * 256 + n] - mean) * rstd * lng[n] + lnb[n];
      outp[(size_t)(row0 + r) * 256 + n] = fmaxf(v, 0.f);
    }
  }
}

// ---------------------------------------------------------------------------
extern "C" void kernel_launch(void* const* d_in, const int* in_sizes, int n_in,
                              void* d_out, int out_size, void* d_ws, size_t ws_size,
                              hipStream_t stream)
{
  (void)in_sizes; (void)n_in; (void)out_size; (void)ws_size;

  const float* x    = (const float*)d_in[0];
  const int*   eidx = (const int*)  d_in[1];
  const float* w0n  = (const float*)d_in[2];
  const float* w0r  = (const float*)d_in[3];
  const float* b0   = (const float*)d_in[4];
  const float* w1n  = (const float*)d_in[5];
  const float* w1r  = (const float*)d_in[6];
  const float* b1   = (const float*)d_in[7];
  const float* ln0g = (const float*)d_in[8];
  const float* ln0b = (const float*)d_in[9];
  const float* ln1g = (const float*)d_in[10];
  const float* ln1b = (const float*)d_in[11];
  const float* linw = (const float*)d_in[12];
  const float* linb = (const float*)d_in[13];
  const int* src = eidx;
  const int* dst = eidx + NE;

  // Workspace carve-up (~130 MB total)
  char* ws = (char*)d_ws; size_t off = 0;
  auto carve = [&](size_t bytes) -> void* {
    off = (off + 255) & ~(size_t)255;
    void* p = ws + off; off += bytes; return p;
  };
  float*  deg   = (float*) carve((size_t)NN * 4);
  float*  inv   = (float*) carve((size_t)NN * 4);
  float*  agg0  = (float*) carve((size_t)NN * CIN  * 4);
  float*  agg1  = (float*) carve((size_t)NN * COUT * 4);
  float*  ybuf  = (float*) carve((size_t)NN * COUT * 4);
  __bf16* B0hi  = (__bf16*)carve((size_t)COUT * 256 * 2);
  __bf16* B0lo  = (__bf16*)carve((size_t)COUT * 256 * 2);
  __bf16* B1hi  = (__bf16*)carve((size_t)COUT * 640 * 2);
  __bf16* B1lo  = (__bf16*)carve((size_t)COUT * 640 * 2);
  float*  bias0 = (float*) carve(COUT * 4);
  float*  bias1 = (float*) carve(COUT * 4);

  hipMemsetAsync(deg,  0, (size_t)NN * 4,            stream);
  hipMemsetAsync(agg0, 0, (size_t)NN * CIN  * 4,     stream);
  hipMemsetAsync(agg1, 0, (size_t)NN * COUT * 4,     stream);

  prep_weights<<<(COUT * 640 + 255) / 256, 256, 0, stream>>>(
      w0n, w0r, w1n, w1r, linw, b0, b1, linb,
      B0hi, B0lo, B1hi, B1lo, bias0, bias1);

  deg_kernel<<<(NE + 255) / 256, 256, 0, stream>>>(dst, deg);
  invdeg_kernel<<<(NN + 255) / 256, 256, 0, stream>>>(deg, inv);

  // conv0 scatter: x -> agg0
  scatter_kernel<CIN><<<NE / 8, 256, 0, stream>>>(x, src, dst, agg0);

  // stage 0: y = relu(LN([agg0*inv | x] @ B0 + b0))
  gemm_ln_relu<256><<<NN / 16, 256, 0, stream>>>(
      agg0, inv, CIN, x, CIN, (const float*)nullptr, 0,
      B0hi, B0lo, bias0, ln0g, ln0b, ybuf);

  // conv1 scatter: y -> agg1
  scatter_kernel<COUT><<<NE / 8, 256, 0, stream>>>(ybuf, src, dst, agg1);

  // stage 1: out = relu(LN([agg1*inv | y | x] @ B1 + b1 + blin))
  gemm_ln_relu<640><<<NN / 16, 256, 0, stream>>>(
      agg1, inv, COUT, ybuf, COUT, x, CIN,
      B1hi, B1lo, bias1, ln1g, ln1b, (float*)d_out);
}
